// Attention_66949950210549
// MI455X (gfx1250) — matrix operand order repaired and checked by reading, not compiled, not run on hardware
//
#include <hip/hip_runtime.h>
#include <hip/hip_bf16.h>

// ---- problem constants (match reference) ----
#define Bq   2
#define Nq   2048
#define Cq   1024
#define Hh   16
#define HDq  64
#define NIN  1024            // N/2, rope positions restart here
#define LOGC (-0.6931471805599453f)   // log(0.5)
#define SCL  (0.125f)                 // 64^-0.5

typedef __attribute__((ext_vector_type(16))) _Float16 v16h;
typedef __attribute__((ext_vector_type(4)))  _Float16 v4h;
typedef __attribute__((ext_vector_type(8)))  float    v8f;
typedef __attribute__((ext_vector_type(4)))  int      i32x4;

union FragH { v16h h; unsigned int u[8]; };

__device__ __forceinline__ v8f wmma32(v16h a, v16h b, v8f c) {
  // D(16x16,f32) = A(16x32,f16) * B(32x16,f16) + C
  return __builtin_amdgcn_wmma_f32_16x16x32_f16(false, a, false, b,
                                                (short)0, c, false, false);
}

// A-fragment k order used here (contiguous, k = 16*hi + s): VGPR v holds k-pair base:
__device__ __forceinline__ int a_kpair(int v, int hi) {
  return 2 * v + hi * 16;   // even; pair (k,k+1)
}

// Stage a 64x32 fp32 tile into LDS as f16: float4 loads, half4 LDS stores.
__device__ __forceinline__ void stage_tile_f32(
    const float* __restrict__ src, int stride, _Float16 (*dst)[32], int tid)
{
  for (int i = tid; i < 512; i += 128) {      // 512 quads, 4 per thread
    const int r = i >> 3, k4 = (i & 7) * 4;
    const float4 f = *(const float4*)(src + (size_t)r * stride + k4);
    v4h hv = { (_Float16)f.x, (_Float16)f.y, (_Float16)f.z, (_Float16)f.w };
    *(v4h*)&dst[r][k4] = hv;
  }
}

// ---- async global->LDS (CDNA5, tracked by ASYNCcnt), guarded ----
#if __has_builtin(__builtin_amdgcn_global_load_async_to_lds_b128) && \
    __has_builtin(__builtin_amdgcn_s_wait_asynccnt)
#define HAVE_ASYNC_LDS 1
#else
#define HAVE_ASYNC_LDS 0
#endif

#if HAVE_ASYNC_LDS
__device__ __forceinline__ void async_cp16(const void* g, void* l) {
  __builtin_amdgcn_global_load_async_to_lds_b128(
      (__attribute__((address_space(1))) i32x4*)(void*)g,
      (__attribute__((address_space(3))) i32x4*)l, 0, 0);
}
#endif

// ====================================================================
// Kernel 1: qkv = x @ qkv_w^T, fused per-head LayerNorm (q,k) + RoPE.
// Block tile 64 rows x 64 cols (one (s,head) per column tile).
// q,k -> f16 (B,H,N,HD);  v -> f16 transposed (B,H,HD,N)
// ====================================================================
__global__ __launch_bounds__(128) void qkv_ln_rope_kernel(
    const float* __restrict__ x, const float* __restrict__ w,
    const float* __restrict__ qn_w, const float* __restrict__ qn_b,
    const float* __restrict__ kn_w, const float* __restrict__ kn_b,
    _Float16* __restrict__ qh, _Float16* __restrict__ kh,
    _Float16* __restrict__ vt)
{
  __shared__ _Float16 lsA[64][32];
  __shared__ _Float16 lsB[64][32];
  const int tid  = threadIdx.x;
  const int wave = tid >> 5;
  const int lane = tid & 31;
  const int hi   = lane >> 4;
  const int lc   = lane & 15;
  const int row0 = blockIdx.x * 64;   // row in [0, B*N)
  const int col0 = blockIdx.y * 64;   // col in [0, 3*C)

  v8f acc[4] = {};
  for (int k0 = 0; k0 < Cq; k0 += 32) {
    stage_tile_f32(x + (size_t)row0 * Cq + k0, Cq, lsA, tid);
    stage_tile_f32(w + (size_t)col0 * Cq + k0, Cq, lsB, tid);
    if (k0 + 32 < Cq) {
      __builtin_prefetch(&x[(size_t)(row0 + (tid >> 1)) * Cq + k0 + 32], 0, 1);
      __builtin_prefetch(&w[(size_t)(col0 + (tid >> 1)) * Cq + k0 + 32], 0, 1);
    }
    __syncthreads();
    const unsigned int* pA = (const unsigned int*)lsA;
    const unsigned int* pB = (const unsigned int*)lsB;
    FragH fa;
    const int ar = wave * 16 + lc;
#pragma unroll
    for (int v = 0; v < 8; ++v)
      fa.u[v] = pA[ar * 16 + (a_kpair(v, hi) >> 1)];
#pragma unroll
    for (int t = 0; t < 4; ++t) {
      FragH fb;
      const int c = t * 16 + lc;
#pragma unroll
      for (int v = 0; v < 8; ++v)
        fb.u[v] = pB[c * 16 + v + hi * 8];   // k = 2v + 16*hi
      acc[t] = wmma32(fa.h, fb.h, acc[t]);
    }
    __syncthreads();
  }

  const int s = col0 >> 10;            // 0=q 1=k 2=v
  const int h = (col0 & 1023) >> 6;    // head

  if (s < 2) {
    // LayerNorm over HD=64 per row. Row m = v + 8*hi, col = t*16 + lc.
    float sum[8], sq[8];
#pragma unroll
    for (int v = 0; v < 8; ++v) { sum[v] = 0.f; sq[v] = 0.f; }
#pragma unroll
    for (int t = 0; t < 4; ++t)
#pragma unroll
      for (int v = 0; v < 8; ++v) { float a = acc[t][v]; sum[v] += a; sq[v] += a * a; }
#pragma unroll
    for (int off = 8; off >= 1; off >>= 1)
#pragma unroll
      for (int v = 0; v < 8; ++v) {
        sum[v] += __shfl_xor(sum[v], off, 32);
        sq[v]  += __shfl_xor(sq[v],  off, 32);
      }
    const float* lw = (s == 0) ? qn_w : kn_w;
    const float* lb = (s == 0) ? qn_b : kn_b;
    float ln[4][8];
#pragma unroll
    for (int t = 0; t < 4; ++t) {
      const int d = t * 16 + lc;
      const float gw = lw[d], gb = lb[d];
#pragma unroll
      for (int v = 0; v < 8; ++v) {
        const float mu  = sum[v] * (1.f / 64.f);
        const float var = sq[v] * (1.f / 64.f) - mu * mu;
        ln[t][v] = (acc[t][v] - mu) * rsqrtf(var + 1e-5f) * gw + gb;
      }
    }
    _Float16* dst = (s == 0) ? qh : kh;
#pragma unroll
    for (int t = 0; t < 4; ++t) {
      const int d = t * 16 + lc;
      const int j = d & 31;
      const float inv = __powf(10000.f, -(float)(2 * j) * (1.f / 64.f));
#pragma unroll
      for (int v = 0; v < 8; ++v) {
        const int gr  = row0 + wave * 16 + v + 8 * hi;
        const int n   = gr & (Nq - 1);
        const int b   = gr >> 11;
        const int pos = n & (NIN - 1);
        float sn, cs;
        __sincosf((float)pos * inv, &sn, &cs);
        const float rot = (t < 2) ? -ln[t + 2][v] : ln[t - 2][v];
        dst[(size_t)((b * Hh + h) * Nq + n) * HDq + d] =
            (_Float16)(ln[t][v] * cs + rot * sn);
      }
    }
  } else {
    // v: store transposed (B,H,HD,N) for contiguous PV B-fragments
#pragma unroll
    for (int t = 0; t < 4; ++t) {
      const int d = t * 16 + lc;
#pragma unroll
      for (int v = 0; v < 8; ++v) {
        const int gr = row0 + wave * 16 + v + 8 * hi;
        const int n  = gr & (Nq - 1);
        const int b  = gr >> 11;
        vt[(size_t)((b * Hh + h) * HDq + d) * Nq + n] = (_Float16)acc[t][v];
      }
    }
  }
}

// ====================================================================
// Kernel 2: flash attention per (b,h). K/V tiles staged in LDS
// (double-buffered async global->LDS when available). Each wave owns
// 16 q rows; online softmax; P transposed via per-wave LDS scratch.
// ====================================================================
#if HAVE_ASYNC_LDS
__device__ __forceinline__ void issue_kv_tile(
    const _Float16* kp, const _Float16* vp, int kt,
    _Float16 (*dK)[64], _Float16 (*dV)[32], int tid)
{
  // K tile: 32 rows x 64 f16 = 8 x b128 per row; 2 async ops / thread
  for (int i = tid; i < 256; i += 128) {
    int r = i >> 3, c = i & 7;
    async_cp16(kp + (size_t)(kt + r) * HDq + c * 8, &dK[r][c * 8]);
  }
  // V tile: 64 d-rows x 32 f16 = 4 x b128 per row; 2 async ops / thread
  for (int i = tid; i < 256; i += 128) {
    int d = i >> 2, c = i & 3;
    async_cp16(vp + (size_t)d * Nq + kt + c * 8, &dV[d][c * 8]);
  }
}
#endif

__global__ __launch_bounds__(128) void attn_kernel(
    const _Float16* __restrict__ qh, const _Float16* __restrict__ kh,
    const _Float16* __restrict__ vt, _Float16* __restrict__ oh)
{
#if HAVE_ASYNC_LDS
  __shared__ _Float16 lsK[2][32][64];
  __shared__ _Float16 lsV[2][64][32];
#else
  __shared__ _Float16 lsK[1][32][64];
  __shared__ _Float16 lsV[1][64][32];
#endif
  __shared__ _Float16 lsP[4][16][32];   // per-wave P scratch (rows x keys)
  const int tid  = threadIdx.x;
  const int wave = tid >> 5;
  const int lane = tid & 31;
  const int hi   = lane >> 4;
  const int lc   = lane & 15;
  const int bh   = blockIdx.y;
  const int b    = bh / Hh, h = bh % Hh;
  const int q0   = blockIdx.x * 64 + wave * 16;

  const _Float16* qp = qh + (size_t)bh * Nq * HDq;
  const _Float16* kp = kh + (size_t)bh * Nq * HDq;
  const _Float16* vp = vt + (size_t)bh * HDq * Nq;

  // Q as two A-fragments (d 0..31, 32..63)
  FragH qa[2];
  {
    const unsigned int* qrow = (const unsigned int*)(qp + (size_t)(q0 + lc) * HDq);
#pragma unroll
    for (int fi = 0; fi < 2; ++fi)
#pragma unroll
      for (int v = 0; v < 8; ++v)
        qa[fi].u[v] = qrow[(a_kpair(v, hi) + fi * 32) >> 1];
  }

  v8f ot[4] = {};
  float mrow[8], lrow[8];
#pragma unroll
  for (int v = 0; v < 8; ++v) { mrow[v] = -1e30f; lrow[v] = 0.f; }

#if HAVE_ASYNC_LDS
  issue_kv_tile(kp, vp, 0, lsK[0], lsV[0], tid);   // prologue: tile 0
#endif

  for (int kt = 0; kt < Nq; kt += 32) {
#if HAVE_ASYNC_LDS
    const int pb = (kt >> 5) & 1;
    if (kt + 32 < Nq) {
      issue_kv_tile(kp, vp, kt + 32, lsK[pb ^ 1], lsV[pb ^ 1], tid);
      // 4 async ops just issued per wave; <=4 left means current tile landed
      __builtin_amdgcn_s_wait_asynccnt(4);
    } else {
      __builtin_amdgcn_s_wait_asynccnt(0);
    }
    __syncthreads();
    const unsigned int* K32 = (const unsigned int*)lsK[pb];  // [32][32] dwords
    const unsigned int* V32 = (const unsigned int*)lsV[pb];  // [64][16] dwords
#else
    {
      unsigned int* dK = (unsigned int*)lsK[0];
      unsigned int* dV = (unsigned int*)lsV[0];
      for (int i = tid; i < 32 * 32; i += 128) {
        int r = i >> 5, c = i & 31;
        dK[r * 32 + c] = ((const unsigned int*)(kp + (size_t)(kt + r) * HDq))[c];
      }
      for (int i = tid; i < 64 * 16; i += 128) {
        int d = i >> 4, c = i & 15;
        dV[d * 16 + c] = ((const unsigned int*)(vp + (size_t)d * Nq + kt))[c];
      }
    }
    __syncthreads();
    const unsigned int* K32 = (const unsigned int*)lsK[0];
    const unsigned int* V32 = (const unsigned int*)lsV[0];
#endif

    // ---- scores: two 16-key C fragments ----
    v8f sc[2];
#pragma unroll
    for (int hk = 0; hk < 2; ++hk) {
      const int kl = hk * 16 + lc;     // key within tile
      v8f c = {};
#pragma unroll
      for (int fi = 0; fi < 2; ++fi) {
        FragH kb;
#pragma unroll
        for (int v = 0; v < 8; ++v)
          kb.u[v] = K32[kl * 32 + v + hi * 8 + fi * 16]; // d = 2v+16*hi+32*fi
        c = wmma32(qa[fi].h, kb.h, c);
      }
      sc[hk] = c;
    }
    // ---- scale + cross-block bias, row max ----
    float p[2][8], mt[8];
#pragma unroll
    for (int v = 0; v < 8; ++v) mt[v] = -1e30f;
#pragma unroll
    for (int hk = 0; hk < 2; ++hk) {
      const int key = kt + hk * 16 + lc;
      const bool kin = key < NIN;
#pragma unroll
      for (int v = 0; v < 8; ++v) {
        const int qr = q0 + v + 8 * hi;
        const float sval = sc[hk][v] * SCL + (((qr < NIN) != kin) ? LOGC : 0.f);
        p[hk][v] = sval;
        mt[v] = fmaxf(mt[v], sval);
      }
    }
#pragma unroll
    for (int off = 8; off >= 1; off >>= 1)
#pragma unroll
      for (int v = 0; v < 8; ++v)
        mt[v] = fmaxf(mt[v], __shfl_xor(mt[v], off, 32));

    float corr[8], rs[8];
#pragma unroll
    for (int v = 0; v < 8; ++v) {
      const float mn = fmaxf(mrow[v], mt[v]);
      corr[v] = __expf(mrow[v] - mn);
      mrow[v] = mn;
      rs[v] = 0.f;
    }
#pragma unroll
    for (int hk = 0; hk < 2; ++hk)
#pragma unroll
      for (int v = 0; v < 8; ++v) {
        const float e = __expf(p[hk][v] - mrow[v]);
        p[hk][v] = e;
        rs[v] += e;
      }
#pragma unroll
    for (int off = 8; off >= 1; off >>= 1)
#pragma unroll
      for (int v = 0; v < 8; ++v)
        rs[v] += __shfl_xor(rs[v], off, 32);
#pragma unroll
    for (int v = 0; v < 8; ++v) lrow[v] = lrow[v] * corr[v] + rs[v];

    // ---- transpose P (C-layout -> A-layout) through per-wave LDS ----
#pragma unroll
    for (int hk = 0; hk < 2; ++hk)
#pragma unroll
      for (int v = 0; v < 8; ++v)
        lsP[wave][v + 8 * hi][hk * 16 + lc] = (_Float16)p[hk][v];
    asm volatile("s_wait_dscnt 0" ::: "memory");
    FragH pa;
    const unsigned int* pP = (const unsigned int*)lsP[wave];
#pragma unroll
    for (int v = 0; v < 8; ++v)
      pa.u[v] = pP[lc * 16 + (a_kpair(v, hi) >> 1)];

    // ---- rescale O, accumulate P @ V ----
#pragma unroll
    for (int t = 0; t < 4; ++t) {
      const int d = t * 16 + lc;
      FragH vb;
#pragma unroll
      for (int v = 0; v < 8; ++v)
        vb.u[v] = V32[d * 16 + v + hi * 8];    // k-dim key = 2v + 16*hi
      v8f o = ot[t];
#pragma unroll
      for (int v = 0; v < 8; ++v) o[v] *= corr[v];
      ot[t] = wmma32(pa.h, vb.h, o);
    }
    __syncthreads();   // protect staged tiles from next iteration's writes
  }

  // ---- normalize, store o (B,N,C) f16 ----
#pragma unroll
  for (int t = 0; t < 4; ++t) {
    const int d = t * 16 + lc;
#pragma unroll
    for (int v = 0; v < 8; ++v) {
      const int n = q0 + v + 8 * hi;
      oh[(size_t)(b * Nq + n) * Cq + h * HDq + d] =
          (_Float16)(ot[t][v] / lrow[v]);
    }
  }
}

// ====================================================================
// Kernel 3: out = o @ proj_w^T + proj_b   (fp32 output)
// A tile (already f16) staged via async global->LDS when available.
// ====================================================================
__global__ __launch_bounds__(128) void proj_kernel(
    const _Float16* __restrict__ a, const float* __restrict__ w,
    const float* __restrict__ bias, float* __restrict__ out)
{
  __shared__ _Float16 lsA[64][32];
  __shared__ _Float16 lsB[64][32];
  const int tid  = threadIdx.x;
  const int wave = tid >> 5;
  const int lane = tid & 31;
  const int hi   = lane >> 4;
  const int lc   = lane & 15;
  const int row0 = blockIdx.x * 64;
  const int col0 = blockIdx.y * 64;

  v8f acc[4] = {};
  for (int k0 = 0; k0 < Cq; k0 += 32) {
#if HAVE_ASYNC_LDS
    // A tile: 64 rows x 32 f16 (64B) = 4 x b128 per row; async copy
    for (int i = tid; i < 256; i += 128) {
      int r = i >> 2, c = i & 3;
      async_cp16(a + (size_t)(row0 + r) * Cq + k0 + c * 8, &lsA[r][c * 8]);
    }
#else
    {
      unsigned int* lsA32 = (unsigned int*)lsA;
      const unsigned int* a32 = (const unsigned int*)a;
      for (int i = tid; i < 64 * 16; i += 128) {
        int r = i >> 4, c = i & 15;
        lsA32[r * 16 + c] = a32[(size_t)(row0 + r) * (Cq / 2) + (k0 >> 1) + c];
      }
    }
#endif
    stage_tile_f32(w + (size_t)col0 * Cq + k0, Cq, lsB, tid);
    if (k0 + 32 < Cq)
      __builtin_prefetch(&w[(size_t)(col0 + (tid >> 1)) * Cq + k0 + 32], 0, 1);
#if HAVE_ASYNC_LDS
    __builtin_amdgcn_s_wait_asynccnt(0);
#endif
    __syncthreads();
    const unsigned int* pA = (const unsigned int*)lsA;
    const unsigned int* pB = (const unsigned int*)lsB;
    FragH fa;
    const int ar = wave * 16 + lc;
#pragma unroll
    for (int v = 0; v < 8; ++v)
      fa.u[v] = pA[ar * 16 + (a_kpair(v, hi) >> 1)];
#pragma unroll
    for (int t = 0; t < 4; ++t) {
      FragH fb;
      const int c = t * 16 + lc;
#pragma unroll
      for (int v = 0; v < 8; ++v)
        fb.u[v] = pB[c * 16 + v + hi * 8];
      acc[t] = wmma32(fa.h, fb.h, acc[t]);
    }
    __syncthreads();
  }
#pragma unroll
  for (int t = 0; t < 4; ++t) {
    const int c = col0 + t * 16 + lc;
    const float bb = bias[c];
#pragma unroll
    for (int v = 0; v < 8; ++v) {
      const int r = row0 + wave * 16 + v + 8 * hi;
      out[(size_t)r * Cq + c] = acc[t][v] + bb;
    }
  }
}

// ====================================================================
extern "C" void kernel_launch(void* const* d_in, const int* in_sizes, int n_in,
                              void* d_out, int out_size, void* d_ws, size_t ws_size,
                              hipStream_t stream) {
  const float* x      = (const float*)d_in[0];
  const float* qkv_w  = (const float*)d_in[1];
  const float* qn_w   = (const float*)d_in[2];
  const float* qn_b   = (const float*)d_in[3];
  const float* kn_w   = (const float*)d_in[4];
  const float* kn_b   = (const float*)d_in[5];
  const float* proj_w = (const float*)d_in[6];
  const float* proj_b = (const float*)d_in[7];
  float* out = (float*)d_out;

  const size_t per = (size_t)Bq * Hh * Nq * HDq;   // 4M f16 elems
  _Float16* qh = (_Float16*)d_ws;
  _Float16* kh = qh + per;
  _Float16* vt = kh + per;
  _Float16* oh = vt + per;                          // B*N*C f16

  // K1: (B*N)/64 x (3C)/64 blocks
  qkv_ln_rope_kernel<<<dim3(64, 48), 128, 0, stream>>>(
      x, qkv_w, qn_w, qn_b, kn_w, kn_b, qh, kh, vt);
  // K2: N/64 x (B*H) blocks
  attn_kernel<<<dim3(32, 32), 128, 0, stream>>>(qh, kh, vt, oh);
  // K3: (B*N)/64 x C/64 blocks
  proj_kernel<<<dim3(64, 16), 128, 0, stream>>>(oh, proj_w, proj_b, out);
}
